// AdjModel_65283502899535
// MI455X (gfx1250) — matrix operand order repaired and checked by reading, not compiled, not run on hardware
//
#include <hip/hip_runtime.h>
#include <hip/hip_bf16.h>

// ---------------------------------------------------------------------------
// AdjModel (GAT + GRU + min-cost-flow) for MI455X / gfx1250, wave32 + WMMA.
// All per-node 64x64 matmuls run as M16 x N64 x K64 tiles on
// v_wmma_f32_16x16x32_f16 (fp32 accumulate). A and B operands are staged in
// LDS PRE-SWIZZLED into the exact WMMA fragment layout, so each lane loads its
// 16-element fragment as one contiguous 32B LDS read (2x ds_load_b128) instead
// of 16 scattered ds_load_u16 + v_mov_b16 packing.
// ---------------------------------------------------------------------------

#define DEV static __device__ __forceinline__

typedef __attribute__((ext_vector_type(16))) _Float16 v16h;
typedef __attribute__((ext_vector_type(8)))  float    v8f;

constexpr int Bb = 4, Vv = 16384, Kk = 16, Dd = 64;
constexpr float NEGV = -1e9f;

DEV v8f zero8() {
  v8f z;
#pragma unroll
  for (int i = 0; i < 8; ++i) z[i] = 0.0f;
  return z;
}

// Fragment slot for A element (row m, k) of a 16x64 f16 A operand.
// v_wmma_f32_16x16x32_f16 A layout (16-bit, 16x32 per k-step):
//   lane = m + 16*((k>>3)&1);  element j = (k&7) + 8*(k>=16)   (k within step)
DEV int a_frag_idx(int row, int k) {
  int ks = k >> 5, kk = k & 31;
  int lane = row + (((kk >> 3) & 1) << 4);
  int j = (kk & 7) + ((kk & 16) ? 8 : 0);
  return (ks * 32 + lane) * 16 + j;
}

// Load a [nrows x 64] fp32 weight matrix from global into LDS as f16,
// pre-swizzled into B-fragment layout: frag (ks, nt) for lane n holds
// W[ks*32 + kk][nt*16 + (lane&15)], kk = 8*(lane>>4) + (e&7) + 16*(e>=8), e = 0..15.
// Rows >= nrows are zero-padded (encoder input weight has 34 rows).
DEV void load_weights_swz(const float* g, _Float16* s, int nrows) {
  for (int e = threadIdx.x; e < 64 * 64; e += blockDim.x) {
    int r = e >> 6, c = e & 63;
    float v = (r < nrows) ? g[e] : 0.0f;   // coalesced global read
    int ks = r >> 5, kk = r & 31;
    int half = (kk >> 3) & 1, ke = (kk & 7) + ((kk & 16) ? 8 : 0);
    int nt = c >> 4, n = c & 15;
    int lane = (half << 4) | n;
    s[((ks * 4 + nt) * 32 + lane) * 16 + ke] = (_Float16)v;
  }
}

// Copy 16 consecutive node rows (16*64 contiguous floats, coalesced) into an
// LDS A-fragment tile (f16, swizzled).
DEV void fill_tileA(const float* g, int node0, _Float16* sAf) {
  const int lane = threadIdx.x & 31;
#pragma unroll
  for (int i = 0; i < 32; ++i) {
    int e = i * 32 + lane;
    sAf[a_frag_idx(e >> 6, e & 63)] = (_Float16)g[(size_t)node0 * 64 + e];
  }
}

// One wave computes D[16x64] += A[16x64] @ W[64x64], both operands already in
// WMMA fragment layout in LDS. 2 k-steps x 4 n-tiles = 8 v_wmma.
// EXEC must be all 1s here: callers keep control flow block-uniform.
DEV void mma_tile(const _Float16* sAf, const _Float16* sWsw, v8f acc[4]) {
  const int lane = threadIdx.x & 31;
#pragma unroll
  for (int ks = 0; ks < 2; ++ks) {
    v16h a = *(const v16h*)(sAf + (ks * 32 + lane) * 16);
#pragma unroll
    for (int nt = 0; nt < 4; ++nt) {
      v16h b = *(const v16h*)(sWsw + ((ks * 4 + nt) * 32 + lane) * 16);
      acc[nt] = __builtin_amdgcn_wmma_f32_16x16x32_f16(
          false, a, false, b, (short)0, acc[nt], false, false);
    }
  }
}

// ---------------------------------------------------------------------------
// Encoder: h = tanh(tanh([emb|feat|0] @ W1 + b1) @ W2 + b2)
// ---------------------------------------------------------------------------
__global__ void encoder_kernel(const float* __restrict__ emb,
                               const float* __restrict__ feat,
                               const float* __restrict__ W1, const float* __restrict__ b1,
                               const float* __restrict__ W2, const float* __restrict__ b2,
                               float* __restrict__ h) {
  __shared__ alignas(32) _Float16 sW1[64 * 64];
  __shared__ alignas(32) _Float16 sW2[64 * 64];
  __shared__ alignas(32) _Float16 sAf[4][16 * 64];
  __shared__ float sb1[64], sb2[64];
  load_weights_swz(W1, sW1, 34);
  load_weights_swz(W2, sW2, 64);
  for (int e = threadIdx.x; e < 64; e += blockDim.x) { sb1[e] = b1[e]; sb2[e] = b2[e]; }

  const int w = threadIdx.x >> 5, lane = threadIdx.x & 31;
  const int node0 = blockIdx.x * 64 + w * 16;
  // Assemble x = [emb(32) | feat(2) | zeros(30)] straight into fragment layout.
#pragma unroll
  for (int i = 0; i < 32; ++i) {
    int e = i * 32 + lane;
    int r = e >> 6, c = e & 63;
    int nn = node0 + r;
    float v = 0.0f;
    if (c < 32)      v = emb[(size_t)nn * 32 + c];
    else if (c < 34) v = feat[(size_t)nn * 2 + (c - 32)];
    sAf[w][a_frag_idx(r, c)] = (_Float16)v;
  }
  __syncthreads();

  v8f acc[4];
#pragma unroll
  for (int i = 0; i < 4; ++i) acc[i] = zero8();
  mma_tile(sAf[w], sW1, acc);
  __syncthreads();

  // h1 = tanh(acc + b1), written back in A-fragment layout for matmul 2.
  const int n = lane & 15, mo = (lane >> 4) << 3;
#pragma unroll
  for (int nt = 0; nt < 4; ++nt)
#pragma unroll
    for (int r = 0; r < 8; ++r) {
      int col = nt * 16 + n;
      sAf[w][a_frag_idx(mo + r, col)] = (_Float16)tanhf(acc[nt][r] + sb1[col]);
    }
  __syncthreads();

#pragma unroll
  for (int i = 0; i < 4; ++i) acc[i] = zero8();
  mma_tile(sAf[w], sW2, acc);
#pragma unroll
  for (int nt = 0; nt < 4; ++nt)
#pragma unroll
    for (int r = 0; r < 8; ++r) {
      int col = nt * 16 + n;
      h[(size_t)(node0 + mo + r) * 64 + col] = tanhf(acc[nt][r] + sb2[col]);
    }
}

// ---------------------------------------------------------------------------
// GAT head transform: t = h @ W_gat[hd];  tsrc = t . a_src[hd]
// ---------------------------------------------------------------------------
__global__ void gat_transform_kernel(const float* __restrict__ h,
                                     const float* __restrict__ Wg,
                                     const float* __restrict__ asrc,
                                     float* __restrict__ t,
                                     float* __restrict__ tsrc) {
  __shared__ alignas(32) _Float16 sW[64 * 64];
  __shared__ alignas(32) _Float16 sAf[4][16 * 64];
  __shared__ float sas[64];
  load_weights_swz(Wg, sW, 64);
  for (int e = threadIdx.x; e < 64; e += blockDim.x) sas[e] = asrc[e];

  const int w = threadIdx.x >> 5, lane = threadIdx.x & 31;
  const int node0 = blockIdx.x * 64 + w * 16;
  fill_tileA(h, node0, sAf[w]);
  __syncthreads();

  v8f acc[4];
#pragma unroll
  for (int i = 0; i < 4; ++i) acc[i] = zero8();
  mma_tile(sAf[w], sW, acc);

  const int n = lane & 15, mo = (lane >> 4) << 3;
  float dotp[8];
#pragma unroll
  for (int r = 0; r < 8; ++r) dotp[r] = 0.0f;
#pragma unroll
  for (int nt = 0; nt < 4; ++nt)
#pragma unroll
    for (int r = 0; r < 8; ++r) {
      int col = nt * 16 + n;
      float v = acc[nt][r];
      t[(size_t)(node0 + mo + r) * 64 + col] = v;
      dotp[r] += v * sas[col];
    }
  // Reduce over the 16 column-lanes sharing each row.
#pragma unroll
  for (int r = 0; r < 8; ++r) {
    float s = dotp[r];
    s += __shfl_xor(s, 1, 32);
    s += __shfl_xor(s, 2, 32);
    s += __shfl_xor(s, 4, 32);
    s += __shfl_xor(s, 8, 32);
    if (n == 0) tsrc[node0 + mo + r] = s;
  }
}

// ---------------------------------------------------------------------------
// GAT gather + masked softmax + aggregation. One wave per node.
// lane = (k, half): lane k<16 covers dims 0..31, lane k+16 covers dims 32..63.
// ---------------------------------------------------------------------------
__global__ void gat_aggregate_kernel(const float* __restrict__ t,
                                     const float* __restrict__ tsrc,
                                     const float* __restrict__ adst,
                                     const int* __restrict__ adj,
                                     float* __restrict__ agg,
                                     int head_first) {
  __shared__ float snb[4][16 * 64];
  __shared__ float sattn[4][16];
  __shared__ float sad[64];
  for (int e = threadIdx.x; e < 64; e += blockDim.x) sad[e] = adst[e];
  __syncthreads();

  const int w = threadIdx.x >> 5, lane = threadIdx.x & 31;
  const int nn = blockIdx.x * 4 + w;     // flat node over B*V
  const int b = nn >> 14;                // nn / V
  const int k = lane & 15, half = lane >> 4;

  int idx = adj[(size_t)nn * Kk + k];
  bool masked = (idx == Vv);
  const float* row = t + ((size_t)(b * Vv + (masked ? 0 : idx))) * 64 + half * 32;
  __builtin_prefetch(row, 0, 0);         // global_prefetch_b8 the neighbor row

  float sc = 0.0f;
#pragma unroll
  for (int j = 0; j < 32; ++j) {
    float v = masked ? 0.0f : row[j];
    snb[w][k * 64 + half * 32 + j] = v;
    sc += v * sad[half * 32 + j];
  }
  sc += __shfl_xor(sc, 16, 32);          // combine the two halves of the dot

  float score = masked ? NEGV : (tsrc[nn] + sc);
  float mx = score;
  mx = fmaxf(mx, __shfl_xor(mx, 1, 32));
  mx = fmaxf(mx, __shfl_xor(mx, 2, 32));
  mx = fmaxf(mx, __shfl_xor(mx, 4, 32));
  mx = fmaxf(mx, __shfl_xor(mx, 8, 32));
  float ex = __expf(score - mx);
  float sm = ex;
  sm += __shfl_xor(sm, 1, 32);
  sm += __shfl_xor(sm, 2, 32);
  sm += __shfl_xor(sm, 4, 32);
  sm += __shfl_xor(sm, 8, 32);
  float at = ex / sm;
  if (half == 0) sattn[w][k] = at;
  __syncthreads();

  // Each lane produces 2 output dims: weighted sum over the 16 neighbors.
  const int d = lane * 2;
  float s0 = 0.0f, s1 = 0.0f;
#pragma unroll
  for (int kk = 0; kk < 16; ++kk) {
    float a = sattn[w][kk];
    s0 += a * snb[w][kk * 64 + d];
    s1 += a * snb[w][kk * 64 + d + 1];
  }
  float* ap = agg + (size_t)nn * 64 + d;
  if (head_first) { ap[0] = s0;          ap[1] = s1; }
  else            { ap[0] = ap[0] + s0;  ap[1] = ap[1] + s1; }
}

// m = tanh(agg / H), in place
__global__ void mtransform_kernel(float* __restrict__ agg, int n) {
  int i = blockIdx.x * blockDim.x + threadIdx.x;
  if (i < n) agg[i] = tanhf(agg[i] * 0.25f);
}

// ---------------------------------------------------------------------------
// GRU gate: g = sigmoid(m@W + h@U + b); mode==0 -> out=g (z), mode==1 -> out=g*h (rh)
// ---------------------------------------------------------------------------
__global__ void gru_gate_kernel(const float* __restrict__ m, const float* __restrict__ h,
                                const float* __restrict__ W, const float* __restrict__ U,
                                const float* __restrict__ bias,
                                float* __restrict__ outbuf, int mode) {
  __shared__ alignas(32) _Float16 sW[64 * 64];
  __shared__ alignas(32) _Float16 sU[64 * 64];
  __shared__ alignas(32) _Float16 sMf[4][16 * 64];
  __shared__ alignas(32) _Float16 sHf[4][16 * 64];
  __shared__ float sb[64];
  load_weights_swz(W, sW, 64);
  load_weights_swz(U, sU, 64);
  for (int e = threadIdx.x; e < 64; e += blockDim.x) sb[e] = bias[e];

  const int w = threadIdx.x >> 5, lane = threadIdx.x & 31;
  const int node0 = blockIdx.x * 64 + w * 16;
  fill_tileA(m, node0, sMf[w]);
  fill_tileA(h, node0, sHf[w]);
  __syncthreads();

  v8f acc[4];
#pragma unroll
  for (int i = 0; i < 4; ++i) acc[i] = zero8();
  mma_tile(sMf[w], sW, acc);
  mma_tile(sHf[w], sU, acc);

  const int n = lane & 15, mo = (lane >> 4) << 3;
#pragma unroll
  for (int nt = 0; nt < 4; ++nt)
#pragma unroll
    for (int r = 0; r < 8; ++r) {
      int col = nt * 16 + n;
      size_t gi = (size_t)(node0 + mo + r) * 64 + col;
      float g = 1.0f / (1.0f + __expf(-(acc[nt][r] + sb[col])));
      if (mode) g *= h[gi];        // r-gate: emit r*h directly
      outbuf[gi] = g;
    }
}

// c = tanh(m@Wc + rh@Uc + bc);  h = (1-z)*h + z*c   (in place)
__global__ void gru_update_kernel(const float* __restrict__ m, const float* __restrict__ rh,
                                  const float* __restrict__ z,
                                  const float* __restrict__ Wc, const float* __restrict__ Uc,
                                  const float* __restrict__ bc,
                                  float* __restrict__ h) {
  __shared__ alignas(32) _Float16 sW[64 * 64];
  __shared__ alignas(32) _Float16 sU[64 * 64];
  __shared__ alignas(32) _Float16 sMf[4][16 * 64];
  __shared__ alignas(32) _Float16 sRf[4][16 * 64];
  __shared__ float sb[64];
  load_weights_swz(Wc, sW, 64);
  load_weights_swz(Uc, sU, 64);
  for (int e = threadIdx.x; e < 64; e += blockDim.x) sb[e] = bc[e];

  const int w = threadIdx.x >> 5, lane = threadIdx.x & 31;
  const int node0 = blockIdx.x * 64 + w * 16;
  fill_tileA(m, node0, sMf[w]);
  fill_tileA(rh, node0, sRf[w]);
  __syncthreads();

  v8f acc[4];
#pragma unroll
  for (int i = 0; i < 4; ++i) acc[i] = zero8();
  mma_tile(sMf[w], sW, acc);
  mma_tile(sRf[w], sU, acc);

  const int n = lane & 15, mo = (lane >> 4) << 3;
#pragma unroll
  for (int nt = 0; nt < 4; ++nt)
#pragma unroll
    for (int r = 0; r < 8; ++r) {
      int col = nt * 16 + n;
      size_t gi = (size_t)(node0 + mo + r) * 64 + col;
      float c  = tanhf(acc[nt][r] + sb[col]);
      float zz = z[gi];
      h[gi] = (1.0f - zz) * h[gi] + zz * c;
    }
}

// ---------------------------------------------------------------------------
// Decoder: node_w = tanh(h@W1 + b1) . W2 + b2
// ---------------------------------------------------------------------------
__global__ void decoder_kernel(const float* __restrict__ h,
                               const float* __restrict__ W1, const float* __restrict__ b1,
                               const float* __restrict__ W2col, const float* __restrict__ b2,
                               float* __restrict__ nodew) {
  __shared__ alignas(32) _Float16 sW[64 * 64];
  __shared__ alignas(32) _Float16 sAf[4][16 * 64];
  __shared__ float sb[64], sw2[64];
  load_weights_swz(W1, sW, 64);
  for (int e = threadIdx.x; e < 64; e += blockDim.x) { sb[e] = b1[e]; sw2[e] = W2col[e]; }

  const int w = threadIdx.x >> 5, lane = threadIdx.x & 31;
  const int node0 = blockIdx.x * 64 + w * 16;
  fill_tileA(h, node0, sAf[w]);
  __syncthreads();

  v8f acc[4];
#pragma unroll
  for (int i = 0; i < 4; ++i) acc[i] = zero8();
  mma_tile(sAf[w], sW, acc);

  const int n = lane & 15, mo = (lane >> 4) << 3;
  float dotp[8];
#pragma unroll
  for (int r = 0; r < 8; ++r) dotp[r] = 0.0f;
#pragma unroll
  for (int nt = 0; nt < 4; ++nt)
#pragma unroll
    for (int r = 0; r < 8; ++r) {
      int col = nt * 16 + n;
      dotp[r] += tanhf(acc[nt][r] + sb[col]) * sw2[col];
    }
  float bb2 = b2[0];
#pragma unroll
  for (int r = 0; r < 8; ++r) {
    float s = dotp[r];
    s += __shfl_xor(s, 1, 32);
    s += __shfl_xor(s, 2, 32);
    s += __shfl_xor(s, 4, 32);
    s += __shfl_xor(s, 8, 32);
    if (n == 0) nodew[node0 + mo + r] = s + bb2;
  }
}

// Per-node softmax over K gathered node weights (2 nodes per wave, 16 lanes each)
__global__ void edge_softmax_kernel(const float* __restrict__ nodew,
                                    const int* __restrict__ adj,
                                    float* __restrict__ normw) {
  const int tid = threadIdx.x, lane = tid & 31;
  const int nn = blockIdx.x * 8 + (tid >> 5) * 2 + (lane >> 4);
  const int b = nn >> 14;
  const int k = lane & 15;
  int idx = adj[(size_t)nn * Kk + k];
  bool masked = (idx == Vv);
  float s = masked ? NEGV : nodew[b * Vv + idx];
  float mx = s;
  mx = fmaxf(mx, __shfl_xor(mx, 1, 32));
  mx = fmaxf(mx, __shfl_xor(mx, 2, 32));
  mx = fmaxf(mx, __shfl_xor(mx, 4, 32));
  mx = fmaxf(mx, __shfl_xor(mx, 8, 32));
  float ex = __expf(s - mx);
  float sm = ex;
  sm += __shfl_xor(sm, 1, 32);
  sm += __shfl_xor(sm, 2, 32);
  sm += __shfl_xor(sm, 4, 32);
  sm += __shfl_xor(sm, 8, 32);
  normw[(size_t)nn * Kk + k] = ex / sm;
}

// ---------------------------------------------------------------------------
// Flow solver kernels
// ---------------------------------------------------------------------------
__global__ void flow0_kernel(const float* __restrict__ normw, const float* __restrict__ dem,
                             float* __restrict__ f) {
  int i = blockIdx.x * blockDim.x + threadIdx.x;   // over B*V*K
  int nn = i >> 4;
  f[i] = normw[i] * fmaxf(-dem[nn], 0.0f);
}

__global__ void scatter_kernel(const float* __restrict__ f, const int* __restrict__ adj,
                               float* __restrict__ inflow) {
  int i = blockIdx.x * blockDim.x + threadIdx.x;
  int b = i >> 18;                                  // V*K = 2^18 edges per batch
  atomicAdd(&inflow[b * (Vv + 1) + adj[i]], f[i]);
}

__global__ void flow_update_kernel(const float* __restrict__ inflow,
                                   const float* __restrict__ dem,
                                   const float* __restrict__ normw,
                                   float* __restrict__ fnew) {
  int i = blockIdx.x * blockDim.x + threadIdx.x;
  int nn = i >> 4;
  int b = nn >> 14, v = nn & (Vv - 1);
  fnew[i] = normw[i] * fmaxf(inflow[b * (Vv + 1) + v] - dem[nn], 0.0f);
}

__global__ void finalize_kernel(const float* __restrict__ fcur, const float* __restrict__ fprev,
                                float* __restrict__ out_flow, float* __restrict__ out_cost,
                                float* __restrict__ out_pflow) {
  int i = blockIdx.x * blockDim.x + threadIdx.x;
  float fc = fcur[i];
  out_flow[i]  = fc;
  out_pflow[i] = fprev[i];
  float c = fc * fc;
  c += __shfl_xor(c, 1, 32);
  c += __shfl_xor(c, 2, 32);
  c += __shfl_xor(c, 4, 32);
  c += __shfl_xor(c, 8, 32);
  c += __shfl_xor(c, 16, 32);
  if ((threadIdx.x & 31) == 0) atomicAdd(&out_cost[i >> 18], c);
}

// ---------------------------------------------------------------------------
// Host launcher. Needs ws_size >= ~77 MB of float scratch.
// ---------------------------------------------------------------------------
extern "C" void kernel_launch(void* const* d_in, const int* in_sizes, int n_in,
                              void* d_out, int out_size, void* d_ws, size_t ws_size,
                              hipStream_t stream) {
  (void)in_sizes; (void)n_in; (void)out_size; (void)ws_size;

  const float* emb   = (const float*)d_in[0];
  const float* feat  = (const float*)d_in[1];
  const float* dem   = (const float*)d_in[2];
  const int*   adj   = (const int*)  d_in[3];
  // d_in[4] = flow_indices (unused; reference computes inflow via scatter-add)
  // d_in[5] = num_nodes (compile-time constant V)
  const float* Wenc1 = (const float*)d_in[6];
  const float* benc1 = (const float*)d_in[7];
  const float* Wenc2 = (const float*)d_in[8];
  const float* benc2 = (const float*)d_in[9];
  const float* Wgat  = (const float*)d_in[10];
  const float* asrc  = (const float*)d_in[11];
  const float* adst  = (const float*)d_in[12];
  const float* Wz    = (const float*)d_in[13];
  const float* Uz    = (const float*)d_in[14];
  const float* bz    = (const float*)d_in[15];
  const float* Wr    = (const float*)d_in[16];
  const float* Ur    = (const float*)d_in[17];
  const float* br    = (const float*)d_in[18];
  const float* Wc    = (const float*)d_in[19];
  const float* Uc    = (const float*)d_in[20];
  const float* bc    = (const float*)d_in[21];
  const float* Wd1   = (const float*)d_in[22];
  const float* bd1   = (const float*)d_in[23];
  const float* Wd2   = (const float*)d_in[24];
  const float* bd2   = (const float*)d_in[25];

  const int nodes = Bb * Vv;          // 65536
  const int edges = nodes * Kk;       // 1048576
  const int nfeat = nodes * Dd;       // 4194304

  float* ws = (float*)d_ws;
  float* h      = ws;                 // [B*V*64]
  float* t      = ws + (size_t)nfeat;         // head transform, reused as z-buffer
  float* agg    = t  + (size_t)nfeat;         // aggregated messages / m
  float* rhbuf  = agg + (size_t)nfeat;
  float* tsrc   = rhbuf + (size_t)nfeat;      // [B*V]
  float* nodew  = tsrc + (size_t)nodes;       // [B*V]
  float* inflow = nodew + (size_t)nodes;      // [B*(V+1)]
  float* fa     = inflow + (size_t)(Bb * (Vv + 1));
  float* fb     = fa + (size_t)edges;

  float* out       = (float*)d_out;
  float* out_flow  = out;                       // [B*V*K]
  float* out_cost  = out + (size_t)edges;       // [B]
  float* out_normw = out_cost + Bb;             // [B*V*K]
  float* out_pflow = out_normw + (size_t)edges; // [B*V*K]

  // ---- encoder ----
  encoder_kernel<<<nodes / 64, 128, 0, stream>>>(emb, feat, Wenc1, benc1, Wenc2, benc2, h);

  // ---- graph layers ----
  for (int l = 0; l < 2; ++l) {
    for (int hd = 0; hd < 4; ++hd) {
      gat_transform_kernel<<<nodes / 64, 128, 0, stream>>>(
          h, Wgat + (size_t)hd * 64 * 64, asrc + hd * 64, t, tsrc);
      gat_aggregate_kernel<<<nodes / 4, 128, 0, stream>>>(
          t, tsrc, adst + hd * 64, adj, agg, hd == 0 ? 1 : 0);
    }
    mtransform_kernel<<<nfeat / 256, 256, 0, stream>>>(agg, nfeat);
    gru_gate_kernel<<<nodes / 64, 128, 0, stream>>>(agg, h, Wz, Uz, bz, t,     0); // z -> t
    gru_gate_kernel<<<nodes / 64, 128, 0, stream>>>(agg, h, Wr, Ur, br, rhbuf, 1); // rh
    gru_update_kernel<<<nodes / 64, 128, 0, stream>>>(agg, rhbuf, t, Wc, Uc, bc, h);
  }

  // ---- decoder + edge softmax ----
  decoder_kernel<<<nodes / 64, 128, 0, stream>>>(h, Wd1, bd1, Wd2, bd2, nodew);
  edge_softmax_kernel<<<nodes / 8, 128, 0, stream>>>(nodew, adj, out_normw);

  // ---- flow iterations (scan keeps (f_i, f_{i-1})) ----
  flow0_kernel<<<edges / 256, 256, 0, stream>>>(out_normw, dem, fa);
  float* cur = fa; float* prev = fb;
  for (int it = 0; it < 10; ++it) {
    hipMemsetAsync(inflow, 0, sizeof(float) * (size_t)Bb * (Vv + 1), stream);
    scatter_kernel<<<edges / 256, 256, 0, stream>>>(cur, adj, inflow);
    flow_update_kernel<<<edges / 256, 256, 0, stream>>>(inflow, dem, out_normw, prev);
    float* tmp = cur; cur = prev; prev = tmp;   // cur = f_i, prev = f_{i-1}
  }

  hipMemsetAsync(out_cost, 0, sizeof(float) * Bb, stream);
  finalize_kernel<<<edges / 256, 256, 0, stream>>>(cur, prev, out_flow, out_cost, out_pflow);
}
